// Text2Emoji_8589934592384
// MI455X (gfx1250) — matrix operand order repaired and checked by reading, not compiled, not run on hardware
//
#include <hip/hip_runtime.h>
#include <hip/hip_bf16.h>

// ---------------------------------------------------------------------------
// Types for CDNA5 WMMA (wave32): D(16x16 f32) = A(16x32 f16) x B(32x16 f16) + C
// ---------------------------------------------------------------------------
typedef __attribute__((ext_vector_type(16))) _Float16 v16h;
typedef __attribute__((ext_vector_type(8)))  _Float16 v8h;
typedef __attribute__((ext_vector_type(8)))  float    v8f;

__device__ __forceinline__ v8f wmma_f16(v16h a, v16h b, v8f c) {
    return __builtin_amdgcn_wmma_f32_16x16x32_f16(
        /*neg_a=*/false, a, /*neg_b=*/false, b,
        /*c_mod=*/(short)0, c, /*reuse_a=*/false, /*reuse_b=*/false);
}

__device__ __forceinline__ float sigf(float x) { return 1.0f / (1.0f + __expf(-x)); }

// Model dimensions
#define SS 128
#define BB 32
#define TT 48
#define EE 512
#define HH 256
#define DEV 16000

// ---------------------------------------------------------------------------
// f32 -> f16 conversion (weights, activations)
// ---------------------------------------------------------------------------
__global__ void cvt_f32_f16_kernel(const float* __restrict__ s, _Float16* __restrict__ d, size_t n) {
    size_t stride = (size_t)gridDim.x * blockDim.x;
    for (size_t i = (size_t)blockIdx.x * blockDim.x + threadIdx.x; i < n; i += stride)
        d[i] = (_Float16)s[i];
}

// ---------------------------------------------------------------------------
// Embedding gather + f16 convert.
// out[d0][d1][e] = emb[tok[...]][e];  token index transposed for the encoder
// (source_sent is [S,B] but the scan treats B as time).
// ---------------------------------------------------------------------------
__global__ void gather_embed_f16_kernel(const int* __restrict__ tok, const float* __restrict__ emb,
                                        _Float16* __restrict__ out, int D0, int D1, int E, int transpose) {
    size_t total = (size_t)D0 * D1 * E;
    size_t stride = (size_t)gridDim.x * blockDim.x;
    for (size_t idx = (size_t)blockIdx.x * blockDim.x + threadIdx.x; idx < total; idx += stride) {
        int e = (int)(idx % E);
        size_t r = idx / E;
        int d1 = (int)(r % D1);
        int d0 = (int)(r / D1);
        int ti = transpose ? (d1 * D0 + d0) : (d0 * D1 + d1);
        int tk = tok[ti];
        out[idx] = (_Float16)emb[(size_t)tk * E + e];
    }
}

// ---------------------------------------------------------------------------
// Generic WMMA GEMM: C[M,N] = A[M,K] * Bw[N,K]^T (+ bias[N]), fp32 accum.
// One 16x16 tile per wave, grid-strided. M%16==0, N%16==0, K%32==0.
// A-fragment per lane l: row = l%16, 16 contiguous K at (l/16)*16.
// B-fragment per lane l: col = l%16, 16 contiguous K at (l/16)*16 (row of Bw).
// ---------------------------------------------------------------------------
__global__ __launch_bounds__(256) void gemm_f16w_kernel(
    const _Float16* __restrict__ A, int lda,
    const _Float16* __restrict__ Bw, int ldb,
    const float* __restrict__ bias,
    float* __restrict__ C, long long ldc,
    int M, int N, int K)
{
    int lane = threadIdx.x & 31;
    int lo = lane & 15, hi = lane >> 4;
    int wave   = blockIdx.x * (blockDim.x >> 5) + (threadIdx.x >> 5);
    int nWaves = gridDim.x * (blockDim.x >> 5);
    int tm = M >> 4, tn = N >> 4;
    int total = tm * tn;
    for (int tile = wave; tile < total; tile += nWaves) {
        int mt = tile % tm;
        int nt = tile / tm;
        int mrow = mt * 16 + lo;
        int ncol = nt * 16 + lo;
        v8f acc = {};
        for (int kb = 0; kb < K; kb += 32) {
            const _Float16* ap = A + (size_t)mrow * lda + kb + (hi << 4);
            v8h a0 = *(const v8h*)(ap);
            v8h a1 = *(const v8h*)(ap + 8);
            v16h a;
            #pragma unroll
            for (int q = 0; q < 8; ++q) { a[q] = a0[q]; a[q + 8] = a1[q]; }
            v16h b = *(const v16h*)(Bw + (size_t)ncol * ldb + kb + (hi << 4));
            acc = wmma_f16(a, b, acc);
        }
        #pragma unroll
        for (int i = 0; i < 8; ++i) {
            int r = mt * 16 + (hi << 3) + i;
            float v = acc[i] + (bias ? bias[ncol] : 0.0f);
            C[(size_t)r * ldc + ncol] = v;
        }
    }
}

// ---------------------------------------------------------------------------
// Persistent bi-GRU direction scan. One workgroup, 16 waves, 32 time steps.
// h (128x256) lives in LDS as f16 (WMMA A operand). GI = x@Wih^T + b_ih is
// precomputed. Per step: gh = h@Whh^T (+b_hh), gates, h update; h also written
// to the next layer's input buffer at column offset colOff.
// ---------------------------------------------------------------------------
__global__ __launch_bounds__(512) void gru_scan_kernel(
    const float* __restrict__ GI,     // [32][128][768]
    const _Float16* __restrict__ Whh, // [768][256]
    const float* __restrict__ b_hh,   // [768]
    _Float16* __restrict__ out_xs,    // [32][128][512]
    int colOff, int reverse)
{
    __shared__ _Float16 hbuf[SS * HH]; // 64 KB
    int tid = threadIdx.x;
    int wave = tid >> 5, lane = tid & 31;
    int hi = lane >> 4, lo = lane & 15;
    for (int i = tid; i < SS * HH; i += 512) hbuf[i] = (_Float16)0.0f;
    __syncthreads();

    for (int st = 0; st < 32; ++st) {
        int t = reverse ? (31 - st) : st;
        float hnew[8][8];
        #pragma unroll
        for (int j = 0; j < 8; ++j) {
            int ht = wave * 8 + j;       // 128 h-tiles: 8 m-tiles x 16 c-tiles
            int mt = ht >> 4, ct = ht & 15;
            int m = mt * 16 + lo;
            v8f acc[3] = {};
            for (int kb = 0; kb < HH; kb += 32) {
                const _Float16* hrow = hbuf + m * HH + kb + (hi << 4);
                v16h a;
                #pragma unroll
                for (int q = 0; q < 8; ++q) { a[q] = hrow[q]; a[q + 8] = hrow[8 + q]; }
                #pragma unroll
                for (int g = 0; g < 3; ++g) {
                    int n = g * HH + ct * 16 + lo;
                    v16h b = *(const v16h*)(Whh + (size_t)n * HH + kb + (hi << 4));
                    acc[g] = wmma_f16(a, b, acc[g]);
                }
            }
            #pragma unroll
            for (int i = 0; i < 8; ++i) {
                int m2 = mt * 16 + (hi << 3) + i;
                int c  = ct * 16 + lo;
                const float* gi = GI + ((size_t)t * SS + m2) * 768;
                float r  = sigf(gi[c]        + acc[0][i] + b_hh[c]);
                float z  = sigf(gi[HH + c]   + acc[1][i] + b_hh[HH + c]);
                float nn = tanhf(gi[2*HH + c] + r * (acc[2][i] + b_hh[2*HH + c]));
                float hold = (float)hbuf[m2 * HH + c];
                hnew[j][i] = (1.0f - z) * nn + z * hold;
            }
        }
        __syncthreads(); // all reads of old h done before any write
        #pragma unroll
        for (int j = 0; j < 8; ++j) {
            int ht = wave * 8 + j;
            int mt = ht >> 4, ct = ht & 15;
            #pragma unroll
            for (int i = 0; i < 8; ++i) {
                int m2 = mt * 16 + (hi << 3) + i;
                int c  = ct * 16 + lo;
                _Float16 hv = (_Float16)hnew[j][i];
                hbuf[m2 * HH + c] = hv;
                out_xs[((size_t)t * SS + m2) * (2 * HH) + colOff + c] = hv;
            }
        }
        __syncthreads();
    }
}

// ---------------------------------------------------------------------------
// Mask + lengths + initial decoder state: state0[b] = enc[b][len[b]].
// ---------------------------------------------------------------------------
__global__ __launch_bounds__(256) void init_state_kernel(
    const int* __restrict__ src, const float* __restrict__ enc,
    float* __restrict__ state, float* __restrict__ maskadd)
{
    __shared__ int lenS[BB];
    int tid = threadIdx.x;
    if (tid < BB) {
        int cnt = 0;
        for (int s = 0; s < SS; ++s) cnt += (src[s * BB + tid] != 0) ? 1 : 0;
        lenS[tid] = cnt - 1;
    }
    __syncthreads();
    for (int idx = tid; idx < BB * HH; idx += 256) {
        int b = idx >> 8, c = idx & 255;
        state[idx] = enc[((size_t)b * SS + lenS[b]) * HH + c];
    }
    for (int idx = tid; idx < BB * SS; idx += 256) {
        int b = idx >> 7, s = idx & 127;
        maskadd[idx] = (src[s * BB + b] != 0) ? 0.0f : -__builtin_inff();
    }
}

// ---------------------------------------------------------------------------
// Decoder GRU cell for one step: 32 blocks x 1 wave, one 16x16 h-tile each.
// gh = state@Whh^T via WMMA (state converted f32->f16 on the fly).
// ---------------------------------------------------------------------------
__global__ __launch_bounds__(32) void dec_cell_kernel(
    const float* __restrict__ GIt,    // [32][768] (incl. b_ih)
    const _Float16* __restrict__ Whh, // [768][256]
    const float* __restrict__ b_hh,
    const float* __restrict__ state,  // [32][256]
    float* __restrict__ h32, _Float16* __restrict__ h16)
{
    int lane = threadIdx.x;
    int blk = blockIdx.x;          // 0..31 -> 2 m-tiles x 16 c-tiles
    int mt = blk >> 4, ct = blk & 15;
    int hi = lane >> 4, lo = lane & 15;
    int m = mt * 16 + lo;
    v8f acc[3] = {};
    for (int kb = 0; kb < HH; kb += 32) {
        const float* hrow = state + m * HH + kb + (hi << 4);
        v16h a;
        #pragma unroll
        for (int q = 0; q < 8; ++q) { a[q] = (_Float16)hrow[q]; a[q + 8] = (_Float16)hrow[8 + q]; }
        #pragma unroll
        for (int g = 0; g < 3; ++g) {
            int n = g * HH + ct * 16 + lo;
            v16h b = *(const v16h*)(Whh + (size_t)n * HH + kb + (hi << 4));
            acc[g] = wmma_f16(a, b, acc[g]);
        }
    }
    #pragma unroll
    for (int i = 0; i < 8; ++i) {
        int m2 = mt * 16 + (hi << 3) + i;
        int c  = ct * 16 + lo;
        const float* gi = GIt + (size_t)m2 * 768;
        float r  = sigf(gi[c]        + acc[0][i] + b_hh[c]);
        float z  = sigf(gi[HH + c]   + acc[1][i] + b_hh[HH + c]);
        float nn = tanhf(gi[2*HH + c] + r * (acc[2][i] + b_hh[2*HH + c]));
        float hold = state[m2 * HH + c];
        float hv = (1.0f - z) * nn + z * hold;
        h32[m2 * HH + c] = hv;
        h16[m2 * HH + c] = (_Float16)hv;
    }
}

// ---------------------------------------------------------------------------
// Bahdanau attention + state update for one step. Block per batch b.
// score[i] = tanh(encatt[b,i,:] + h@w0R^T + b0) . w1 + b1 ; masked softmax;
// state = h + sum_i w[i] * enc[b,i,:].
// ---------------------------------------------------------------------------
__global__ __launch_bounds__(128) void dec_attn_kernel(
    const float* __restrict__ encatt, // [32][128][256] (enc @ w0_left^T, no bias)
    const float* __restrict__ enc,    // [32][128][256]
    const float* __restrict__ h32,    // [32][256]
    const float* __restrict__ w0,     // attn_w0 [256][512] (f32)
    const float* __restrict__ b0,
    const float* __restrict__ w1,     // [256]
    const float* __restrict__ b1,
    const float* __restrict__ maskadd,// [32][128] additive mask
    float* __restrict__ state)        // in/out [32][256]
{
    int b = blockIdx.x, tid = threadIdx.x;
    __shared__ float hl[HH];
    __shared__ float hatt[HH];
    __shared__ float sc[SS];
    __shared__ float red[SS];

    for (int c = tid; c < HH; c += 128) hl[c] = h32[b * HH + c];
    __syncthreads();
    for (int c = tid; c < HH; c += 128) {
        float acc = b0[c];
        const float* wr = w0 + (size_t)c * (2 * HH) + HH; // right half
        for (int k = 0; k < HH; ++k) acc += hl[k] * wr[k];
        hatt[c] = acc;
    }
    __syncthreads();
    {
        float s = b1[0];
        const float* ea = encatt + ((size_t)b * SS + tid) * HH;
        for (int c = 0; c < HH; ++c) s += tanhf(ea[c] + hatt[c]) * w1[c];
        sc[tid] = s + maskadd[b * SS + tid];
    }
    __syncthreads();
    red[tid] = sc[tid]; __syncthreads();
    for (int o = 64; o > 0; o >>= 1) { if (tid < o) red[tid] = fmaxf(red[tid], red[tid + o]); __syncthreads(); }
    float mx = red[0]; __syncthreads();
    float e = __expf(sc[tid] - mx);
    red[tid] = e; __syncthreads();
    for (int o = 64; o > 0; o >>= 1) { if (tid < o) red[tid] += red[tid + o]; __syncthreads(); }
    float inv = 1.0f / red[0]; __syncthreads();
    sc[tid] = e * inv; __syncthreads();
    for (int c = tid; c < HH; c += 128) {
        float acc = 0.0f;
        const float* eb = enc + (size_t)b * SS * HH + c;
        for (int i = 0; i < SS; ++i) acc += sc[i] * eb[(size_t)i * HH];
        state[b * HH + c] = hl[c] + acc;
    }
}

// ---------------------------------------------------------------------------
// Host side
// ---------------------------------------------------------------------------
static void launch_gemm(const _Float16* A, int lda, const _Float16* B, int ldb,
                        const float* bias, float* C, long long ldc,
                        int M, int N, int K, hipStream_t stream)
{
    int tiles = (M / 16) * (N / 16);
    int blocks = (tiles + 7) / 8;  // 8 waves / block
    if (blocks > 4096) blocks = 4096;
    gemm_f16w_kernel<<<blocks, 256, 0, stream>>>(A, lda, B, ldb, bias, C, ldc, M, N, K);
}

static void launch_cvt(const float* s, _Float16* d, size_t n, hipStream_t stream) {
    int blocks = (int)((n + 255) / 256);
    if (blocks > 4096) blocks = 4096;
    cvt_f32_f16_kernel<<<blocks, 256, 0, stream>>>(s, d, n);
}

extern "C" void kernel_launch(void* const* d_in, const int* in_sizes, int n_in,
                              void* d_out, int out_size, void* d_ws, size_t ws_size,
                              hipStream_t stream)
{
    (void)in_sizes; (void)n_in; (void)out_size; (void)ws_size;

    const int*   src       = (const int*)d_in[0];
    const int*   tgt       = (const int*)d_in[1];
    const float* enc_emb   = (const float*)d_in[2];
    const float* gru_w_ih  = (const float*)d_in[3];
    const float* gru_w_hh  = (const float*)d_in[4];
    const float* gru_b_ih  = (const float*)d_in[5];
    const float* gru_b_hh  = (const float*)d_in[6];
    const float* hid_lin_w = (const float*)d_in[7];
    const float* hid_lin_b = (const float*)d_in[8];
    const float* dec_emb   = (const float*)d_in[9];
    const float* cell_w_ih = (const float*)d_in[10];
    const float* cell_w_hh = (const float*)d_in[11];
    const float* cell_b_ih = (const float*)d_in[12];
    const float* cell_b_hh = (const float*)d_in[13];
    const float* out_w     = (const float*)d_in[14];
    const float* out_b     = (const float*)d_in[15];
    const float* attn_w0   = (const float*)d_in[16];
    const float* attn_b0   = (const float*)d_in[17];
    const float* attn_w1   = (const float*)d_in[18];
    const float* attn_b1   = (const float*)d_in[19];

    char* ws = (char*)d_ws;
    size_t off = 0;
    auto alloc = [&](size_t bytes) -> void* {
        void* p = ws + off;
        off = (off + bytes + 255) & ~(size_t)255;
        return p;
    };

    _Float16* w_ih16   = (_Float16*)alloc((size_t)2*2*768*512 * 2);
    _Float16* w_hh16   = (_Float16*)alloc((size_t)2*2*768*256 * 2);
    _Float16* hidlin16 = (_Float16*)alloc((size_t)256*512 * 2);
    _Float16* cwih16   = (_Float16*)alloc((size_t)768*512 * 2);
    _Float16* cwhh16   = (_Float16*)alloc((size_t)768*256 * 2);
    _Float16* outw16   = (_Float16*)alloc((size_t)DEV*256 * 2);
    _Float16* w016     = (_Float16*)alloc((size_t)256*512 * 2);
    _Float16* xsA      = (_Float16*)alloc((size_t)32*128*512 * 2);
    _Float16* xsB      = (_Float16*)alloc((size_t)32*128*512 * 2);
    float*    GI       = (float*)   alloc((size_t)32*128*768 * 4);
    float*    ENC      = (float*)   alloc((size_t)4096*256 * 4);
    _Float16* ENC16    = (_Float16*)alloc((size_t)4096*256 * 2);
    float*    ENCATT   = (float*)   alloc((size_t)4096*256 * 4);
    float*    GIdec    = (float*)   alloc((size_t)1504*768 * 4);
    _Float16* decx16   = (_Float16*)alloc((size_t)1504*512 * 2);
    float*    state    = (float*)   alloc((size_t)BB*HH * 4);
    float*    h32      = (float*)   alloc((size_t)BB*HH * 4);
    _Float16* h16      = (_Float16*)alloc((size_t)BB*HH * 2);
    float*    maskadd  = (float*)   alloc((size_t)BB*SS * 4);

    // 1. Weight conversions (one-time; all weights then sit in f16 / L2).
    launch_cvt(gru_w_ih,  w_ih16,   (size_t)2*2*768*512, stream);
    launch_cvt(gru_w_hh,  w_hh16,   (size_t)2*2*768*256, stream);
    launch_cvt(hid_lin_w, hidlin16, (size_t)256*512,     stream);
    launch_cvt(cell_w_ih, cwih16,   (size_t)768*512,     stream);
    launch_cvt(cell_w_hh, cwhh16,   (size_t)768*256,     stream);
    launch_cvt(out_w,     outw16,   (size_t)DEV*256,     stream);
    launch_cvt(attn_w0,   w016,     (size_t)256*512,     stream);

    // 2. Encoder embedding gather (transposed: scan axis is B).
    gather_embed_f16_kernel<<<4096, 256, 0, stream>>>(src, enc_emb, xsA, 32, 128, 512, 1);

    // 3. Bi-GRU layers: hoisted input GEMM + persistent recurrent scan.
    _Float16* curIn = xsA;
    _Float16* curOut = xsB;
    for (int l = 0; l < 2; ++l) {
        for (int d = 0; d < 2; ++d) {
            const _Float16* wih = w_ih16 + (size_t)(l*2 + d) * 768 * 512;
            const _Float16* whh = w_hh16 + (size_t)(l*2 + d) * 768 * 256;
            const float* bih = gru_b_ih + (l*2 + d) * 768;
            const float* bhh = gru_b_hh + (l*2 + d) * 768;
            launch_gemm(curIn, 512, wih, 512, bih, GI, 768, 4096, 768, 512, stream);
            gru_scan_kernel<<<1, 512, 0, stream>>>(GI, whh, bhh, curOut, d * 256, d);
        }
        _Float16* tmp = curIn; curIn = curOut; curOut = tmp;
    }

    // 4. enc_seq = hid_lin(xs)  -> ENC [B,S,H] row-major (row = b*128 + s).
    launch_gemm(curIn, 512, hidlin16, 512, hid_lin_b, ENC, 256, 4096, 256, 512, stream);

    // 5. Attention encoder projection: ENCATT = ENC @ attn_w0[:, :H]^T.
    launch_cvt(ENC, ENC16, (size_t)4096*256, stream);
    launch_gemm(ENC16, 256, w016, 512, nullptr, ENCATT, 256, 4096, 256, 256, stream);

    // 6. Mask, lengths, initial decoder state.
    init_state_kernel<<<1, 256, 0, stream>>>(src, ENC, state, maskadd);

    // 7. Decoder input-side GEMM for all 47 steps.
    gather_embed_f16_kernel<<<3008, 256, 0, stream>>>(tgt, dec_emb, decx16, 47, 32, 512, 0);
    launch_gemm(decx16, 512, cwih16, 512, cell_b_ih, GIdec, 768, 1504, 768, 512, stream);

    // 8. Decoder loop: cell -> logits (WMMA, scattered into [B,T-1,V]) -> attn.
    float* outp = (float*)d_out;
    for (int t = 0; t < 47; ++t) {
        dec_cell_kernel<<<32, 32, 0, stream>>>(GIdec + (size_t)t * 32 * 768,
                                               cwhh16, cell_b_hh, state, h32, h16);
        launch_gemm(h16, 256, outw16, 256, out_b,
                    outp + (size_t)t * DEV, (long long)47 * DEV,
                    32, DEV, 256, stream);
        dec_attn_kernel<<<32, 128, 0, stream>>>(ENCATT, ENC, h32, attn_w0, attn_b0,
                                                attn_w1, attn_b1, maskadd, state);
    }
}
